// RHDC_39573828665592
// MI455X (gfx1250) — matrix operand order repaired and checked by reading, not compiled, not run on hardware
//
#include <hip/hip_runtime.h>

typedef __attribute__((ext_vector_type(16))) _Float16 v16h;
typedef __attribute__((ext_vector_type(8)))  float    v8f;

#define IN_F 128
#define TILE 16
#define DIFF_T 3

// ---------- utility kernels ----------

__global__ void k_zero_f4(float4* __restrict__ p, int n4) {
    int i = blockIdx.x * blockDim.x + threadIdx.x;
    if (i < n4) { float4 z; z.x = z.y = z.z = z.w = 0.f; p[i] = z; }
}

// counts[r*N + node]; rel 0 starts at 1 (self-loop)
__global__ void k_init_counts(int* __restrict__ counts, int N, int total) {
    int i = blockIdx.x * blockDim.x + threadIdx.x;
    if (i < total) counts[i] = (i < N) ? 1 : 0;
}

__global__ void k_count_edges(const int* __restrict__ dst, const int* __restrict__ et,
                              int* __restrict__ counts, int N, int nE) {
    int e = blockIdx.x * blockDim.x + threadIdx.x;
    if (e < nE) atomicAdd(&counts[et[e] * N + dst[e]], 1);
}

// Pre-swizzle W (fp32 [r][K=128][N=128]) into per-lane WMMA B layout (f16).
// Flat index: (((r*8 + ct)*4 + kc)*32 + lane)*16 + h
// B element: K = kc*32 + (lane/16)*16 + h ; Ncol = ct*16 + (lane%16)
__global__ void k_swizzle_W(const float* __restrict__ W, _Float16* __restrict__ Bsw, int total) {
    int idx = blockIdx.x * blockDim.x + threadIdx.x;
    if (idx >= total) return;
    int h16  = idx & 15;
    int lane = (idx >> 4) & 31;
    int kc   = (idx >> 9) & 3;
    int ct   = (idx >> 11) & 7;
    int r    = idx >> 14;
    int K    = kc * 32 + (lane >> 4) * 16 + h16;
    int Ncol = ct * 16 + (lane & 15);
    Bsw[idx] = (_Float16)W[((size_t)r * IN_F + K) * IN_F + Ncol];
}

// h_next = (rel 0: h_cur (self-loop term)) else 0
__global__ void k_init_next(const float4* __restrict__ hc, float4* __restrict__ hn,
                            int copySelf, int n4) {
    int i = blockIdx.x * blockDim.x + threadIdx.x;
    if (i >= n4) return;
    if (copySelf) hn[i] = hc[i];
    else { float4 z; z.x = z.y = z.z = z.w = 0.f; hn[i] = z; }
}

// One wave (32 lanes) per edge; each lane moves a float4 of the 128-float row.
__global__ void k_scatter(const float* __restrict__ hc, float* __restrict__ hn,
                          const int* __restrict__ src, const int* __restrict__ dst,
                          const int* __restrict__ et, int r, int nE) {
    unsigned tid = blockIdx.x * blockDim.x + threadIdx.x;
    unsigned e = tid >> 5;
    if (e >= (unsigned)nE) return;
    if (et[e] != r) return;               // wave-coherent branch
    int lane = tid & 31;
    const float4 v = ((const float4*)(hc + (size_t)src[e] * IN_F))[lane];
    float* o = hn + (size_t)dst[e] * IN_F + lane * 4;
    atomicAdd(o + 0, v.x);
    atomicAdd(o + 1, v.y);
    atomicAdd(o + 2, v.z);
    atomicAdd(o + 3, v.w);
}

__global__ void k_normalize(float4* __restrict__ hn, const int* __restrict__ counts,
                            int r, int N, int n4) {
    int i = blockIdx.x * blockDim.x + threadIdx.x;
    if (i >= n4) return;
    int node = i >> 5;                    // 32 float4 per node row
    int c = counts[r * N + node];
    float4 v = hn[i];
    if (c > 0) {
        float inv = 1.0f / (float)c;
        v.x *= inv; v.y *= inv; v.z *= inv; v.w *= inv;
    } else {
        v.x = v.y = v.z = v.w = 0.f;
    }
    hn[i] = v;
}

// out[16-row strip] += h_strip @ Bsw_r  (f16 WMMA, f32 accumulate)
// 256 threads = 8 waves; wave w handles column tile w; grid.x = N/16.
__global__ void k_gemm_acc(const float* __restrict__ h, const _Float16* __restrict__ Bsw_r,
                           float* __restrict__ out) {
    const int lane = threadIdx.x & 31;
    const int ct   = threadIdx.x >> 5;    // column tile 0..7
    const int m    = lane & 15;
    const int hi   = lane >> 4;           // lane group 0/1
    const int rowBase = blockIdx.x * TILE;
    const int col  = ct * 16 + m;         // C: N = lane%16

    // C/D layout: VGPR j -> M = hi*8 + j, N = lane%16
    v8f c;
#pragma unroll
    for (int j = 0; j < 8; ++j)
        c[j] = out[(size_t)(rowBase + hi * 8 + j) * IN_F + col];

    // A layout: row M = lane%16; halves 0..15 -> K = k0 + hi*16 + t
    const float* arow = h + (size_t)(rowBase + m) * IN_F;

#pragma unroll
    for (int kc = 0; kc < 4; ++kc) {
        const int k0 = kc * 32;
        const float4* pa = (const float4*)(arow + k0 + hi * 16);
        float4 f0 = pa[0], f1 = pa[1];
        const float4* pb = (const float4*)(arow + k0 + hi * 16 + 8);
        float4 g0 = pb[0], g1 = pb[1];
        v16h a;
        a[0]  = (_Float16)f0.x; a[1]  = (_Float16)f0.y;
        a[2]  = (_Float16)f0.z; a[3]  = (_Float16)f0.w;
        a[4]  = (_Float16)f1.x; a[5]  = (_Float16)f1.y;
        a[6]  = (_Float16)f1.z; a[7]  = (_Float16)f1.w;
        a[8]  = (_Float16)g0.x; a[9]  = (_Float16)g0.y;
        a[10] = (_Float16)g0.z; a[11] = (_Float16)g0.w;
        a[12] = (_Float16)g1.x; a[13] = (_Float16)g1.y;
        a[14] = (_Float16)g1.z; a[15] = (_Float16)g1.w;

        v16h b = *(const v16h*)(Bsw_r + (((ct * 4 + kc) * 32 + lane) << 4));

        c = __builtin_amdgcn_wmma_f32_16x16x32_f16(
                /*neg_a=*/false, a, /*neg_b=*/false, b,
                /*c_mod=*/(short)0, c, /*reuse_a=*/false, /*reuse_b=*/false);
    }

#pragma unroll
    for (int j = 0; j < 8; ++j)
        out[(size_t)(rowBase + hi * 8 + j) * IN_F + col] = c[j];
}

__global__ void k_relu(float* __restrict__ out, int n) {
    int i = blockIdx.x * blockDim.x + threadIdx.x;
    if (i < n) out[i] = fmaxf(out[i], 0.f);
}

// ---------- host-side orchestration ----------

extern "C" void kernel_launch(void* const* d_in, const int* in_sizes, int n_in,
                              void* d_out, int out_size, void* d_ws, size_t ws_size,
                              hipStream_t stream) {
    const float* features = (const float*)d_in[0];
    const float* W        = (const float*)d_in[1];
    const int*   src      = (const int*)d_in[2];
    const int*   dst      = (const int*)d_in[3];
    const int*   et       = (const int*)d_in[4];

    const int N       = in_sizes[0] / IN_F;           // 50000
    const int nE      = in_sizes[2];                  // 600000
    const int NUM_REL = in_sizes[1] / (IN_F * IN_F);  // 6

    // workspace layout
    char* ws = (char*)d_ws;
    int* counts = (int*)ws;
    size_t off = (((size_t)NUM_REL * N * sizeof(int)) + 255) & ~(size_t)255;
    _Float16* Bsw = (_Float16*)(ws + off);
    const int swTotal = NUM_REL * 8 * 4 * 32 * 16;    // halves
    off += (((size_t)swTotal * sizeof(_Float16)) + 255) & ~(size_t)255;
    float* hA = (float*)(ws + off);
    off += (size_t)N * IN_F * sizeof(float);
    float* hB = (float*)(ws + off);

    float* out = (float*)d_out;
    const int n4  = N * IN_F / 4;
    const int TPB = 256;

    k_zero_f4<<<(n4 + TPB - 1) / TPB, TPB, 0, stream>>>((float4*)out, n4);
    k_init_counts<<<(NUM_REL * N + TPB - 1) / TPB, TPB, 0, stream>>>(counts, N, NUM_REL * N);
    k_count_edges<<<(nE + TPB - 1) / TPB, TPB, 0, stream>>>(dst, et, counts, N, nE);
    k_swizzle_W<<<(swTotal + TPB - 1) / TPB, TPB, 0, stream>>>(W, Bsw, swTotal);

    for (int r = 0; r < NUM_REL; ++r) {
        const float* cur = features;
        float* nxt = hA;
        for (int t = 0; t < DIFF_T; ++t) {
            k_init_next<<<(n4 + TPB - 1) / TPB, TPB, 0, stream>>>(
                (const float4*)cur, (float4*)nxt, (r == 0) ? 1 : 0, n4);
            int nThr = nE * 32;   // one wave per edge
            k_scatter<<<(nThr + TPB - 1) / TPB, TPB, 0, stream>>>(
                cur, nxt, src, dst, et, r, nE);
            k_normalize<<<(n4 + TPB - 1) / TPB, TPB, 0, stream>>>(
                (float4*)nxt, counts, r, N, n4);
            cur = nxt;
            nxt = (cur == hA) ? hB : hA;
        }
        k_gemm_acc<<<N / TILE, 256, 0, stream>>>(cur, Bsw + (size_t)r * 8 * 4 * 32 * 16, out);
    }

    k_relu<<<(N * IN_F + TPB - 1) / TPB, TPB, 0, stream>>>(out, N * IN_F);
}
